// AtencionMultiCabeza_23699629540018
// MI455X (gfx1250) — matrix operand order repaired and checked by reading, not compiled, not run on hardware
//
#include <hip/hip_runtime.h>

// ---------------- problem constants ----------------
#define BATCH 4
#define SEQ   2048
#define DMODEL 1024
#define NHEAD 16
#define DKH   64
#define MROWS (BATCH * SEQ)          // 8192

typedef float  v8f  __attribute__((ext_vector_type(8)));
typedef float  v4f  __attribute__((ext_vector_type(4)));
typedef __bf16 v16bf __attribute__((ext_vector_type(16)));
typedef __bf16 v8bf  __attribute__((ext_vector_type(8)));
typedef __bf16 v4bf  __attribute__((ext_vector_type(4)));

union U16BF { v16bf v; v8bf h8[2]; };

// Async global->LDS copy (CDNA5 GLOBAL_LOAD_ASYNC_TO_LDS_B128, ASYNCcnt-tracked).
// Probe-discovered parameter type: pointer to `int __attribute__((vector_size(16)))`.
typedef int v4i_b128 __attribute__((vector_size(16)));
typedef __attribute__((address_space(1))) v4i_b128* glob_b128_ptr;
typedef __attribute__((address_space(3))) v4i_b128* lds_b128_ptr;

#if defined(__has_builtin)
#if __has_builtin(__builtin_amdgcn_global_load_async_to_lds_b128)
#define USE_ASYNC_LDS 1
#endif
#endif
#ifndef USE_ASYNC_LDS
#define USE_ASYNC_LDS 0
#endif

__device__ __forceinline__ void wait_asynccnt0() {
#if defined(__has_builtin) && __has_builtin(__builtin_amdgcn_s_wait_asynccnt)
  __builtin_amdgcn_s_wait_asynccnt(0);
#else
  asm volatile("s_wait_asynccnt 0x0" ::: "memory");
#endif
}

// ---------------- fp32 -> bf16 convert ----------------
__global__ __launch_bounds__(256) void cvt_bf16_kernel(const float* __restrict__ src,
                                                       __bf16* __restrict__ dst, int n4) {
  int i = blockIdx.x * 256 + threadIdx.x;
  if (i < n4) {
    v4f x = *(const v4f*)(src + (size_t)i * 4);
    v4bf y;
    #pragma unroll
    for (int e = 0; e < 4; ++e) y[e] = (__bf16)x[e];
    *(v4bf*)(dst + (size_t)i * 4) = y;
  }
}

// ---------------- GEMM: Y = X @ W^T + b ----------------
// X: [M,1024] bf16 row-major, W: [1024,1024] bf16 row-major (row = output col).
// OUT_F32=false: write bf16 in head-split [B,H,S,DK] layout (fused split_heads).
// OUT_F32=true : write f32 flat [M,1024] (final projection into d_out).
template <bool OUT_F32>
__global__ __launch_bounds__(256) void gemm_wmma_kernel(const __bf16* __restrict__ X,
                                                        const __bf16* __restrict__ W,
                                                        const float*  __restrict__ bias,
                                                        __bf16* __restrict__ Ybf,
                                                        float*  __restrict__ Yf) {
  const int tid  = threadIdx.x;
  const int wave = tid >> 5;
  const int lane = tid & 31;
  const int lr   = lane & 15;
  const int half = lane >> 4;
  const int waveM = wave >> 1;                 // 0..3
  const int waveN = wave & 1;                  // 0..1
  const int mBase = blockIdx.x * 128 + waveM * 32;
  const int nBase = blockIdx.y * 128 + waveN * 64;

  v8f acc[2][4];
  #pragma unroll
  for (int mt = 0; mt < 2; ++mt)
    #pragma unroll
    for (int nt = 0; nt < 4; ++nt)
      #pragma unroll
      for (int e = 0; e < 8; ++e) acc[mt][nt][e] = 0.0f;

  for (int k0 = 0; k0 < DMODEL; k0 += 32) {
    v16bf a[2];
    #pragma unroll
    for (int mt = 0; mt < 2; ++mt) {
      const __bf16* xr = X + (size_t)(mBase + mt * 16 + lr) * DMODEL + k0;
      U16BF u;
      u.h8[0] = *(const v8bf*)(xr + half * 16);         // K = k0 + half*16 .. +7
      u.h8[1] = *(const v8bf*)(xr + half * 16 + 8);     // K = k0+half*16+8 .. +15
      a[mt] = u.v;
    }
    #pragma unroll
    for (int nt = 0; nt < 4; ++nt) {
      // B frag: lane -> N = nBase+nt*16+lr ; elems e -> K = k0 + half*16 + e
      const __bf16* wr = W + (size_t)(nBase + nt * 16 + lr) * DMODEL + k0 + half * 16;
      v16bf bfrag = *(const v16bf*)wr;
      #pragma unroll
      for (int mt = 0; mt < 2; ++mt)
        acc[mt][nt] = __builtin_amdgcn_wmma_f32_16x16x32_bf16(
            false, a[mt], false, bfrag, (short)0, acc[mt][nt], false, false);
    }
  }

  #pragma unroll
  for (int mt = 0; mt < 2; ++mt)
    #pragma unroll
    for (int nt = 0; nt < 4; ++nt)
      #pragma unroll
      for (int r = 0; r < 8; ++r) {
        int row = mBase + mt * 16 + r + 8 * half;
        int col = nBase + nt * 16 + lr;
        float val = acc[mt][nt][r] + bias[col];
        if (OUT_F32) {
          Yf[(size_t)row * DMODEL + col] = val;
        } else {
          int b = row >> 11, s = row & (SEQ - 1);
          int h = col >> 6,  dk = col & (DKH - 1);
          Ybf[(((size_t)(b * NHEAD + h) * SEQ) + s) * DKH + dk] = (__bf16)val;
        }
      }
}

// ---------------- flash attention (causal) ----------------
// grid: (S/64, B*H); block: 128 threads = 4 waves; wave w owns q-rows [w*16, w*16+16).
__global__ __launch_bounds__(128) void flash_attn_kernel(const __bf16* __restrict__ Qh,
                                                         const __bf16* __restrict__ Kh,
                                                         const __bf16* __restrict__ Vh,
                                                         __bf16* __restrict__ attnOut) {
  __shared__ __align__(32) __bf16 Ktile[64 * 64];       // [kv][dk] row-major
  __shared__ __align__(32) __bf16 Vt[64 * 64];          // transposed: [dk][kv]
  __shared__ __align__(32) __bf16 Ptile[4][16 * 64];    // per-wave P staging

  const int tid  = threadIdx.x;
  const int wave = tid >> 5;
  const int lane = tid & 31;
  const int lr   = lane & 15;
  const int half = lane >> 4;

  const int bh = blockIdx.y;                 // b*NHEAD + h
  const int qt = blockIdx.x;
  const int b  = bh >> 4;
  const int h  = bh & (NHEAD - 1);

  const __bf16* Qbase = Qh + ((size_t)bh * SEQ + qt * 64) * DKH;
  const __bf16* Kbase = Kh + (size_t)bh * SEQ * DKH;
  const __bf16* Vbase = Vh + (size_t)bh * SEQ * DKH;

  // Q A-fragments for this wave's 16 rows, both K-steps (dk 0..31, 32..63)
  v16bf qf[2];
  {
    const __bf16* qr = Qbase + (size_t)(wave * 16 + lr) * DKH;
    #pragma unroll
    for (int ks = 0; ks < 2; ++ks) {
      U16BF u;
      u.h8[0] = *(const v8bf*)(qr + ks * 32 + half * 16);
      u.h8[1] = *(const v8bf*)(qr + ks * 32 + half * 16 + 8);
      qf[ks] = u.v;
    }
  }

  v8f o[4];
  float m_i[8], l_i[8];
  #pragma unroll
  for (int nt = 0; nt < 4; ++nt)
    #pragma unroll
    for (int e = 0; e < 8; ++e) o[nt][e] = 0.0f;
  #pragma unroll
  for (int r = 0; r < 8; ++r) { m_i[r] = -1e30f; l_i[r] = 0.0f; }

  for (int kt = 0; kt <= qt; ++kt) {
    const __bf16* Kg = Kbase + (size_t)kt * 64 * DKH;
    const __bf16* Vg = Vbase + (size_t)kt * 64 * DKH;

    // cooperative stage: K row-major (async DMA to LDS), V transposed (via VGPRs)
    #pragma unroll
    for (int i = 0; i < 4; ++i) {
      int idx = (i * 128 + tid) * 8;                    // 0..4095 step 8
#if USE_ASYNC_LDS
      __builtin_amdgcn_global_load_async_to_lds_b128(
          (glob_b128_ptr)(Kg + idx),
          (lds_b128_ptr)(Ktile + idx),
          0, 0);
#else
      *(v8bf*)(Ktile + idx) = *(const v8bf*)(Kg + idx);
#endif
      v8bf vv = *(const v8bf*)(Vg + idx);
      int kv = idx >> 6, dk0 = idx & 63;
      #pragma unroll
      for (int e = 0; e < 8; ++e) Vt[(dk0 + e) * 64 + kv] = vv[e];
    }
    if (kt < qt) {                                      // prefetch next KV tile
      __builtin_prefetch((const void*)(Kg + 64 * DKH + tid * 32), 0, 1);
      __builtin_prefetch((const void*)(Vg + 64 * DKH + tid * 32), 0, 1);
    }
#if USE_ASYNC_LDS
    wait_asynccnt0();                                   // drain this wave's async copies
#endif
    __syncthreads();

    // ---- S = Q K^T ----
    v8f s[4];
    #pragma unroll
    for (int nt = 0; nt < 4; ++nt)
      #pragma unroll
      for (int e = 0; e < 8; ++e) s[nt][e] = 0.0f;
    #pragma unroll
    for (int ks = 0; ks < 2; ++ks)
      #pragma unroll
      for (int nt = 0; nt < 4; ++nt) {
        // B frag: N = kv = nt*16+lr ; K = dk = ks*32 + half*16 + e
        const __bf16* kp = Ktile + (nt * 16 + lr) * 64 + ks * 32 + half * 16;
        v16bf bfrag = *(const v16bf*)kp;
        s[nt] = __builtin_amdgcn_wmma_f32_16x16x32_bf16(
            false, qf[ks], false, bfrag, (short)0, s[nt], false, false);
      }

    // ---- online softmax (rows live in 16-lane groups; reduce via shfl_xor) ----
    const bool diag = (kt == qt);
    #pragma unroll
    for (int r = 0; r < 8; ++r) {
      const int rowLoc = wave * 16 + r + 8 * half;      // q row within 64-block
      float mx = -1e30f;
      #pragma unroll
      for (int nt = 0; nt < 4; ++nt) {
        float x = s[nt][r] * 0.125f;                    // 1/sqrt(64)
        if (diag && (nt * 16 + lr) > rowLoc) x = -1e9f; // causal mask on diagonal tile
        s[nt][r] = x;
        mx = fmaxf(mx, x);
      }
      for (int msk = 1; msk < 16; msk <<= 1) mx = fmaxf(mx, __shfl_xor(mx, msk, 32));
      float mnew = fmaxf(m_i[r], mx);
      float corr = __expf(m_i[r] - mnew);
      float rs = 0.0f;
      #pragma unroll
      for (int nt = 0; nt < 4; ++nt) {
        float p = __expf(s[nt][r] - mnew);
        s[nt][r] = p;
        rs += p;
      }
      for (int msk = 1; msk < 16; msk <<= 1) rs += __shfl_xor(rs, msk, 32);
      l_i[r] = l_i[r] * corr + rs;
      m_i[r] = mnew;
      #pragma unroll
      for (int nt = 0; nt < 4; ++nt) o[nt][r] *= corr;
    }

    // ---- P through LDS (C-layout -> A-layout reshape), then O += P V ----
    __bf16* P = &Ptile[wave][0];
    #pragma unroll
    for (int nt = 0; nt < 4; ++nt)
      #pragma unroll
      for (int r = 0; r < 8; ++r)
        P[(r + 8 * half) * 64 + nt * 16 + lr] = (__bf16)s[nt][r];

    #pragma unroll
    for (int ks = 0; ks < 2; ++ks) {
      U16BF u;                                          // A frag of P: row=lr, K=kv
      const __bf16* pr = P + lr * 64 + ks * 32 + half * 16;
      u.h8[0] = *(const v8bf*)pr;
      u.h8[1] = *(const v8bf*)(pr + 8);
      #pragma unroll
      for (int nt = 0; nt < 4; ++nt) {
        // B frag of V: N = dk = nt*16+lr ; K = kv = ks*32 + half*16 + e (contiguous in Vt)
        const __bf16* vp = Vt + (nt * 16 + lr) * 64 + ks * 32 + half * 16;
        v16bf bfrag = *(const v16bf*)vp;
        o[nt] = __builtin_amdgcn_wmma_f32_16x16x32_bf16(
            false, u.v, false, bfrag, (short)0, o[nt], false, false);
      }
    }
    __syncthreads();                                    // protect K/V tiles for next kt
  }

  // ---- normalize + write [B,S,D] (fused merge-heads) ----
  #pragma unroll
  for (int nt = 0; nt < 4; ++nt)
    #pragma unroll
    for (int r = 0; r < 8; ++r) {
      float val = o[nt][r] / l_i[r];
      int qrow = qt * 64 + wave * 16 + r + 8 * half;
      int dk   = nt * 16 + lr;
      attnOut[((size_t)b * SEQ + qrow) * DMODEL + h * DKH + dk] = (__bf16)val;
    }
}

// ---------------- launch ----------------
extern "C" void kernel_launch(void* const* d_in, const int* in_sizes, int n_in,
                              void* d_out, int out_size, void* d_ws, size_t ws_size,
                              hipStream_t stream) {
  const float* q  = (const float*)d_in[0];
  const float* k  = (const float*)d_in[1];
  const float* v  = (const float*)d_in[2];
  /* d_in[3] = causal mask: handled analytically */
  const float* Wq = (const float*)d_in[4];
  const float* bq = (const float*)d_in[5];
  const float* Wk = (const float*)d_in[6];
  const float* bk = (const float*)d_in[7];
  const float* Wv = (const float*)d_in[8];
  const float* bv = (const float*)d_in[9];
  const float* Wo = (const float*)d_in[10];
  const float* bo = (const float*)d_in[11];
  float* out = (float*)d_out;

  char* ws = (char*)d_ws;
  size_t off = 0;
  auto carve = [&](size_t bytes) {
    off = (off + 255) & ~(size_t)255;
    void* p = ws + off;
    off += bytes;
    return p;
  };
  const size_t BIG = (size_t)MROWS * DMODEL;            // 8M elems
  const size_t WSZ = (size_t)DMODEL * DMODEL;           // 1M elems
  __bf16* qb  = (__bf16*)carve(BIG * 2);
  __bf16* kb  = (__bf16*)carve(BIG * 2);
  __bf16* vb  = (__bf16*)carve(BIG * 2);
  __bf16* wqb = (__bf16*)carve(WSZ * 2);
  __bf16* wkb = (__bf16*)carve(WSZ * 2);
  __bf16* wvb = (__bf16*)carve(WSZ * 2);
  __bf16* wob = (__bf16*)carve(WSZ * 2);
  __bf16* Qh  = (__bf16*)carve(BIG * 2);                // [B,H,S,DK]
  __bf16* Kh  = (__bf16*)carve(BIG * 2);
  __bf16* Vh  = (__bf16*)carve(BIG * 2);
  __bf16* aO  = (__bf16*)carve(BIG * 2);                // [B,S,D]

  // 1) convert fp32 -> bf16
  {
    int n4 = (int)(BIG / 4);
    cvt_bf16_kernel<<<n4 / 256, 256, 0, stream>>>(q, qb, n4);
    cvt_bf16_kernel<<<n4 / 256, 256, 0, stream>>>(k, kb, n4);
    cvt_bf16_kernel<<<n4 / 256, 256, 0, stream>>>(v, vb, n4);
    int w4 = (int)(WSZ / 4);
    cvt_bf16_kernel<<<w4 / 256, 256, 0, stream>>>(Wq, wqb, w4);
    cvt_bf16_kernel<<<w4 / 256, 256, 0, stream>>>(Wk, wkb, w4);
    cvt_bf16_kernel<<<w4 / 256, 256, 0, stream>>>(Wv, wvb, w4);
    cvt_bf16_kernel<<<w4 / 256, 256, 0, stream>>>(Wo, wob, w4);
  }

  // 2) QKV projections (fused bias + split_heads)
  dim3 ggrid(MROWS / 128, DMODEL / 128);
  gemm_wmma_kernel<false><<<ggrid, 256, 0, stream>>>(qb, wqb, bq, Qh, nullptr);
  gemm_wmma_kernel<false><<<ggrid, 256, 0, stream>>>(kb, wkb, bk, Kh, nullptr);
  gemm_wmma_kernel<false><<<ggrid, 256, 0, stream>>>(vb, wvb, bv, Vh, nullptr);

  // 3) causal flash attention
  dim3 agrid(SEQ / 64, BATCH * NHEAD);
  flash_attn_kernel<<<agrid, 128, 0, stream>>>(Qh, Kh, Vh, aO);

  // 4) output projection -> f32 d_out
  gemm_wmma_kernel<true><<<ggrid, 256, 0, stream>>>(aO, wob, bo, nullptr, out);
}
